// GraniteMoeHybridAttentionDecoderLayer_7799660610068
// MI455X (gfx1250) — compile-verified, matrix operand order checked by reading, not run on hardware
//
#include <hip/hip_runtime.h>
#include <hip/hip_bf16.h>

// ---------------------------------------------------------------------------
// MI455X (gfx1250, wave32) implementation of a GraniteMoeHybrid decoder layer.
// All large GEMMs + attention run on v_wmma_f32_16x16x32_bf16 (bf16 in, f32
// accum).  A-activation tiles move global->LDS via the CDNA5 async-DMA path
// (global_load_async_to_lds_b128 + s_wait_asynccnt); f32 weight tiles are
// loaded as b128, converted to bf16, and staged into LDS, double-buffered.
// Workspace requirement: ~75 MiB (see layout below).
// ---------------------------------------------------------------------------

static constexpr int T  = 1024;
static constexpr int D  = 2048;
static constexpr int H  = 32;
static constexpr int KV = 8;
static constexpr int HD = 64;
static constexpr int E  = 8;
static constexpr int I  = 1024;
static constexpr int SI = 4096;
static constexpr float ATTN_MULT = 0.0078125f;
static constexpr float RES_MULT  = 0.22f;
static constexpr float EPS       = 1e-6f;

typedef __bf16 bf16x16 __attribute__((ext_vector_type(16)));
typedef float  f32x8   __attribute__((ext_vector_type(8)));

union BF16Frag {
    unsigned short u[16];
    bf16x16 v;
};

__device__ __forceinline__ unsigned short f2bf(float f) {
    unsigned int u = __float_as_uint(f);
    unsigned int r = u + 0x7FFFu + ((u >> 16) & 1u);   // round-to-nearest-even
    return (unsigned short)(r >> 16);
}

// A/B fragment K-index pattern for V_WMMA_*_16X16X32_BF16 (ISA 7.12.2):
// VGPR p (0..7), half w; lanes 0-15 vs 16-31 differ by +8 in K.
__device__ __forceinline__ int fragK(int e, int lane) {
    int p = e >> 1, w = e & 1;
    return ((p & 3) << 1) + w + ((p >> 2) << 4) + ((lane >> 4) << 3);
}

// Generic pointer -> 32-bit LDS byte offset (AS3) for async-DMA VDST operand.
__device__ __forceinline__ unsigned lds_off(const void* p) {
    return (unsigned)(unsigned long long)
        (__attribute__((address_space(3))) const void*)p;
}

// CDNA5 async global->LDS copy, 16B per lane (ASYNCcnt tracked).
__device__ __forceinline__ void async_copy_b128(unsigned lds_byte_off,
                                                const void* gsrc) {
    asm volatile("global_load_async_to_lds_b128 %0, %1, off"
                 :: "v"(lds_byte_off), "v"(gsrc)
                 : "memory");
}
__device__ __forceinline__ void wait_async0() {
    asm volatile("s_wait_asynccnt 0x0" ::: "memory");
}

// ---------------------------------------------------------------------------
// RMSNorm: one block per row.  Writes f32 and/or bf16 outputs.
// ---------------------------------------------------------------------------
__global__ __launch_bounds__(256)
void rmsnorm_kernel(const float* __restrict__ in, const float* __restrict__ w,
                    float* outf, unsigned short* outb) {
    int row = blockIdx.x;
    int tid = threadIdx.x;
    __shared__ float red[256];
    const float* x = in + (size_t)row * D;
    float ss = 0.f;
    for (int i = tid; i < D; i += 256) { float v = x[i]; ss += v * v; }
    red[tid] = ss;
    __syncthreads();
    for (int s = 128; s > 0; s >>= 1) {
        if (tid < s) red[tid] += red[tid + s];
        __syncthreads();
    }
    float rs = rsqrtf(red[0] / (float)D + EPS);
    for (int i = tid; i < D; i += 256) {
        float v = x[i] * rs * w[i];
        if (outf) outf[(size_t)row * D + i] = v;
        if (outb) outb[(size_t)row * D + i] = f2bf(v);
    }
}

// ---------------------------------------------------------------------------
// RoPE: f32 in -> bf16 out, per (t, head, i<HD/2).
// ---------------------------------------------------------------------------
__global__ __launch_bounds__(256)
void rope_kernel(const int* __restrict__ pos, const float* __restrict__ in,
                 unsigned short* __restrict__ out, int nheads) {
    int idx = blockIdx.x * blockDim.x + threadIdx.x;
    int total = T * nheads * (HD / 2);
    if (idx >= total) return;
    int i = idx & 31;
    int h = (idx >> 5) % nheads;
    int t = idx / (32 * nheads);
    float inv = __expf(-(float)i * (9.2103403719761840f / 32.f)); // 10000^(-i/32)
    float ang = (float)pos[t] * inv;
    float s, c;
    __sincosf(ang, &s, &c);
    size_t base = (size_t)t * nheads * HD + (size_t)h * HD;
    float x1 = in[base + i];
    float x2 = in[base + 32 + i];
    out[base + i]      = f2bf(x1 * c - x2 * s);
    out[base + 32 + i] = f2bf(x2 * c + x1 * s);
}

// ---------------------------------------------------------------------------
// Router: one wave per token.  logits = xn . router_w[e], softmax, top-2.
// ---------------------------------------------------------------------------
__global__ __launch_bounds__(32)
void router_kernel(const float* __restrict__ xn, const float* __restrict__ rw,
                   float* __restrict__ fullw) {
    int t = blockIdx.x;
    int lane = threadIdx.x;
    float acc[E];
#pragma unroll
    for (int e = 0; e < E; ++e) acc[e] = 0.f;
    const float* x = xn + (size_t)t * D;
    for (int i = lane; i < D; i += 32) {
        float xv = x[i];
#pragma unroll
        for (int e = 0; e < E; ++e) acc[e] += xv * rw[(size_t)e * D + i];
    }
#pragma unroll
    for (int e = 0; e < E; ++e)
        for (int off = 16; off > 0; off >>= 1)
            acc[e] += __shfl_xor(acc[e], off, 32);
    if (lane == 0) {
        int i1 = 0;
        for (int e = 1; e < E; ++e) if (acc[e] > acc[i1]) i1 = e;
        int i2 = (i1 == 0) ? 1 : 0;
        for (int e = 0; e < E; ++e) if (e != i1 && acc[e] > acc[i2]) i2 = e;
        float e1 = 1.f;
        float e2 = __expf(acc[i2] - acc[i1]);
        float inv = 1.f / (e1 + e2);
        for (int e = 0; e < E; ++e) {
            float wv = (e == i1) ? e1 * inv : (e == i2) ? e2 * inv : 0.f;
            fullw[(size_t)t * E + e] = wv;
        }
    }
}

// ---------------------------------------------------------------------------
// BF16-WMMA GEMM, double-buffered LDS, async A-tile DMA.
// C[M,N] = A[M,K](bf16) x B[K,N](f32->bf16).
// Block tile 64x128, 8 waves (2M x 4N), each wave 32x32 (4 WMMAs / K-step).
// Requires: M%64==0, N%128==0, K%32==0, lda%8==0, ldb%4==0 (all satisfied).
// ---------------------------------------------------------------------------
#define GEMM_STORE    0  // outf/outb = acc
#define GEMM_RESID    1  // outf = resid + scale*acc
#define GEMM_MOE_INIT 2  // outf  = rowsc[row]*acc
#define GEMM_MOE_ACC  3  // outf += rowsc[row]*acc
#define GEMM_SILU     4  // outb = silu(aux[row,col]) * acc   (bf16)
#define GEMM_FINAL    5  // outf = resid + scale*(aux + acc)

__global__ __launch_bounds__(256)
void gemm_bf16_wmma(const unsigned short* __restrict__ A, int lda,
                    const float* __restrict__ B, int ldb,
                    int M, int N, int K,
                    float* outf, unsigned short* outb, int ldc,
                    int mode, float scale,
                    const float* __restrict__ resid, int ldr,
                    const float* __restrict__ rowsc, int rss,
                    const float* __restrict__ aux, int ldaux) {
    constexpr int AS = 40;   // LDS stride for A rows   [64][AS]  (80B, 16B-mult)
    constexpr int BS = 40;   // LDS stride for B^T rows [128][BS]
    __shared__ __align__(16) unsigned short lds_a[2][64 * AS];
    __shared__ __align__(16) unsigned short lds_b[2][128 * BS];

    int tid  = threadIdx.x;
    int lane = tid & 31;
    int wid  = tid >> 5;
    int wm   = wid >> 2;   // 0..1
    int wn   = wid & 3;    // 0..3
    int m0   = blockIdx.y * 64;
    int n0   = blockIdx.x * 128;
    int ln   = lane & 15;

    // A async staging: 256 threads x b128 covers the 64x32 bf16 tile.
    int ar = tid >> 2;          // row 0..63
    int ac = (tid & 3) << 3;    // col 0,8,16,24

    f32x8 zero = {0.f,0.f,0.f,0.f,0.f,0.f,0.f,0.f};
    f32x8 acc[2][2] = {{zero, zero}, {zero, zero}};

    auto stageA = [&](int buf, int k0) {
        const void* g = &A[(size_t)(m0 + ar) * lda + k0 + ac];
        async_copy_b128(lds_off(&lds_a[buf][ar * AS + ac]), g);
    };
    auto loadB = [&](int k0, float4* r) {
#pragma unroll
        for (int p = 0; p < 4; ++p) {
            int fidx = (tid + p * 256) << 2;        // 0..4095
            int kk = fidx >> 7, n = fidx & 127;
            r[p] = *(const float4*)&B[(size_t)(k0 + kk) * ldb + n0 + n];
        }
    };
    auto storeB = [&](int buf, const float4* r) {
#pragma unroll
        for (int p = 0; p < 4; ++p) {
            int fidx = (tid + p * 256) << 2;
            int kk = fidx >> 7, n = fidx & 127;
            lds_b[buf][(n + 0) * BS + kk] = f2bf(r[p].x);
            lds_b[buf][(n + 1) * BS + kk] = f2bf(r[p].y);
            lds_b[buf][(n + 2) * BS + kk] = f2bf(r[p].z);
            lds_b[buf][(n + 3) * BS + kk] = f2bf(r[p].w);
        }
    };

    // Prologue: stage tile 0.
    stageA(0, 0);
    {
        float4 r[4];
        loadB(0, r);
        storeB(0, r);
    }
    wait_async0();
    __syncthreads();

    int nk = K >> 5;
    int buf = 0;
    for (int kt = 0; kt < nk; ++kt) {
        float4 r[4];
        bool nxt = (kt + 1) < nk;
        if (nxt) {
            stageA(buf ^ 1, (kt + 1) << 5);   // async DMA into other buffer
            loadB((kt + 1) << 5, r);          // b128 weight loads into regs
        }

        BF16Frag af[2], bf[2];
#pragma unroll
        for (int s = 0; s < 2; ++s)
#pragma unroll
            for (int e = 0; e < 16; ++e) {
                int kk = fragK(e, lane);
                af[s].u[e] = lds_a[buf][(wm * 32 + s * 16 + ln) * AS + kk];
                bf[s].u[e] = lds_b[buf][(wn * 32 + s * 16 + ln) * BS + kk];
            }
#pragma unroll
        for (int sm = 0; sm < 2; ++sm)
#pragma unroll
            for (int sn = 0; sn < 2; ++sn)
                acc[sm][sn] = __builtin_amdgcn_wmma_f32_16x16x32_bf16(
                    false, af[sm].v, false, bf[sn].v, (short)0,
                    acc[sm][sn], false, false);

        if (nxt) storeB(buf ^ 1, r);
        wait_async0();
        __syncthreads();
        buf ^= 1;
    }

    int half8 = (lane & 16) >> 1;          // +8 row offset for lanes 16-31
#pragma unroll
    for (int sm = 0; sm < 2; ++sm) {
#pragma unroll
        for (int sn = 0; sn < 2; ++sn) {
            int col = n0 + wn * 32 + sn * 16 + ln;
#pragma unroll
            for (int rr = 0; rr < 8; ++rr) {
                int row = m0 + wm * 32 + sm * 16 + rr + half8;
                if (row >= M || col >= N) continue;
                float v = acc[sm][sn][rr];
                size_t oidx = (size_t)row * ldc + col;
                if (mode == GEMM_STORE) {
                    if (outf) outf[oidx] = v;
                    if (outb) outb[oidx] = f2bf(v);
                } else if (mode == GEMM_RESID) {
                    outf[oidx] = resid[(size_t)row * ldr + col] + scale * v;
                } else if (mode == GEMM_MOE_INIT) {
                    outf[oidx] = rowsc[(size_t)row * rss] * v;
                } else if (mode == GEMM_MOE_ACC) {
                    outf[oidx] += rowsc[(size_t)row * rss] * v;
                } else if (mode == GEMM_SILU) {
                    float g = aux[(size_t)row * ldaux + col];
                    float s = g / (1.f + __expf(-g));
                    outb[oidx] = f2bf(s * v);
                } else { // GEMM_FINAL
                    outf[oidx] = resid[(size_t)row * ldr + col] +
                                 scale * (aux[(size_t)row * ldaux + col] + v);
                }
            }
        }
    }
}

// ---------------------------------------------------------------------------
// Causal GQA flash attention.  One wave per (head, 16-query tile); 4 waves
// per block.  QK^T and P.V via bf16 WMMA; online softmax over 32-key chunks.
// ---------------------------------------------------------------------------
__global__ __launch_bounds__(128)
void attn_kernel(const unsigned short* __restrict__ qb,
                 const unsigned short* __restrict__ kb,
                 const unsigned short* __restrict__ vb,
                 unsigned short* __restrict__ attnb) {
    int h    = blockIdx.x;
    int w    = threadIdx.x >> 5;
    int lane = threadIdx.x & 31;
    int qt   = blockIdx.y * 4 + w;   // 16-query tile index
    int kvh  = h >> 2;               // H/KV = 4
    int hb   = (lane & 16) >> 1;     // +8 row offset for lanes 16-31
    int ln   = lane & 15;

    __shared__ unsigned short pbuf[4][16 * 34];   // per-wave P staging

    BF16Frag aq[2];
#pragma unroll
    for (int c = 0; c < 2; ++c)
#pragma unroll
        for (int e = 0; e < 16; ++e) {
            int kd = c * 32 + fragK(e, lane);
            int qrow = qt * 16 + ln;
            aq[c].u[e] = qb[(size_t)qrow * (H * HD) + h * HD + kd];
        }

    f32x8 o0 = {0,0,0,0,0,0,0,0}, o1 = o0, o2 = o0, o3 = o0;
    float rmax[8], rsum[8];
#pragma unroll
    for (int r = 0; r < 8; ++r) { rmax[r] = -1e30f; rsum[r] = 0.f; }

    int kend = qt * 16 + 16;                      // causal bound
    for (int kbase = 0; kbase < kend; kbase += 32) {
        float p[2][8];
        float tmax[8];
#pragma unroll
        for (int r = 0; r < 8; ++r) tmax[r] = -1e30f;

        for (int sub = 0; sub < 2; ++sub) {
            int ks = kbase + sub * 16;
            f32x8 s = {0,0,0,0,0,0,0,0};
#pragma unroll
            for (int c = 0; c < 2; ++c) {
                BF16Frag bk;
#pragma unroll
                for (int e = 0; e < 16; ++e) {
                    int kd  = c * 32 + fragK(e, lane);
                    int key = ks + ln;
                    bk.u[e] = (key < T)
                        ? kb[(size_t)key * (KV * HD) + kvh * HD + kd] : 0;
                }
                s = __builtin_amdgcn_wmma_f32_16x16x32_bf16(
                    false, aq[c].v, false, bk.v, (short)0, s, false, false);
            }
#pragma unroll
            for (int r = 0; r < 8; ++r) {
                int row = qt * 16 + r + hb;
                int key = ks + ln;
                float sv = s[r] * ATTN_MULT;
                if (key > row || key >= T) sv = -1e30f;
                p[sub][r] = sv;
                tmax[r] = fmaxf(tmax[r], sv);
            }
        }
#pragma unroll
        for (int r = 0; r < 8; ++r)
            for (int off = 8; off > 0; off >>= 1)
                tmax[r] = fmaxf(tmax[r], __shfl_xor(tmax[r], off, 16));

        float corr[8];
#pragma unroll
        for (int r = 0; r < 8; ++r) {
            float mn = fmaxf(rmax[r], tmax[r]);
            corr[r]  = __expf(rmax[r] - mn);
            rmax[r]  = mn;
            p[0][r]  = __expf(p[0][r] - mn);
            p[1][r]  = __expf(p[1][r] - mn);
            float ts = p[0][r] + p[1][r];
            for (int off = 8; off > 0; off >>= 1)
                ts += __shfl_xor(ts, off, 16);
            rsum[r] = rsum[r] * corr[r] + ts;
        }
#pragma unroll
        for (int r = 0; r < 8; ++r) {
            o0[r] *= corr[r]; o1[r] *= corr[r];
            o2[r] *= corr[r]; o3[r] *= corr[r];
        }
#pragma unroll
        for (int sub = 0; sub < 2; ++sub)
#pragma unroll
            for (int r = 0; r < 8; ++r)
                pbuf[w][(r + hb) * 34 + sub * 16 + ln] = f2bf(p[sub][r]);
        BF16Frag pf;
#pragma unroll
        for (int e = 0; e < 16; ++e)
            pf.u[e] = pbuf[w][ln * 34 + fragK(e, lane)];

#pragma unroll
        for (int j = 0; j < 4; ++j) {
            BF16Frag bv;
#pragma unroll
            for (int e = 0; e < 16; ++e) {
                int kd  = fragK(e, lane);
                int key = kbase + kd;
                int n   = j * 16 + ln;
                bv.u[e] = (key < T)
                    ? vb[(size_t)key * (KV * HD) + kvh * HD + n] : 0;
            }
            if (j == 0) o0 = __builtin_amdgcn_wmma_f32_16x16x32_bf16(
                    false, pf.v, false, bv.v, (short)0, o0, false, false);
            if (j == 1) o1 = __builtin_amdgcn_wmma_f32_16x16x32_bf16(
                    false, pf.v, false, bv.v, (short)0, o1, false, false);
            if (j == 2) o2 = __builtin_amdgcn_wmma_f32_16x16x32_bf16(
                    false, pf.v, false, bv.v, (short)0, o2, false, false);
            if (j == 3) o3 = __builtin_amdgcn_wmma_f32_16x16x32_bf16(
                    false, pf.v, false, bv.v, (short)0, o3, false, false);
        }
    }
#pragma unroll
    for (int r = 0; r < 8; ++r) {
        int row = qt * 16 + r + hb;
        float inv = 1.f / rsum[r];
        size_t base = (size_t)row * (H * HD) + h * HD;
        attnb[base +  0 + ln] = f2bf(o0[r] * inv);
        attnb[base + 16 + ln] = f2bf(o1[r] * inv);
        attnb[base + 32 + ln] = f2bf(o2[r] * inv);
        attnb[base + 48 + ln] = f2bf(o3[r] * inv);
    }
}

// ---------------------------------------------------------------------------
// Host-side orchestration.
// ---------------------------------------------------------------------------
static void launch_gemm(hipStream_t s, const unsigned short* A, int lda,
                        const float* B, int ldb, int M, int N, int K,
                        float* outf, unsigned short* outb, int ldc,
                        int mode, float scale,
                        const float* resid, int ldr,
                        const float* rowsc, int rss,
                        const float* aux, int ldaux) {
    dim3 g(N / 128, M / 64);
    gemm_bf16_wmma<<<g, 256, 0, s>>>(A, lda, B, ldb, M, N, K, outf, outb, ldc,
                                     mode, scale, resid, ldr, rowsc, rss,
                                     aux, ldaux);
}

extern "C" void kernel_launch(void* const* d_in, const int* in_sizes, int n_in,
                              void* d_out, int out_size, void* d_ws, size_t ws_size,
                              hipStream_t stream) {
    const int*   pos     = (const int*)  d_in[0];
    const float* hidden  = (const float*)d_in[1];
    const float* ln1     = (const float*)d_in[2];
    const float* ln2     = (const float*)d_in[3];
    const float* wq      = (const float*)d_in[4];
    const float* wk      = (const float*)d_in[5];
    const float* wv      = (const float*)d_in[6];
    const float* wo      = (const float*)d_in[7];
    const float* routerw = (const float*)d_in[8];
    const float* w13     = (const float*)d_in[9];
    const float* w2      = (const float*)d_in[10];
    const float* sw13    = (const float*)d_in[11];
    const float* sw2     = (const float*)d_in[12];

    float* outF     = (float*)d_out;          // final hidden  [T*D]
    float* residOut = outF + (size_t)T * D;   // post-attn residual [T*D]

    char* ws = (char*)d_ws;
    const size_t MB = 1u << 20;
    unsigned short* xb    = (unsigned short*)(ws + 0);        // T*D bf16   4M
    float*          qf    = (float*)         (ws + 4  * MB);  // T*2048 f32 8M
    float*          kf    = (float*)         (ws + 12 * MB);  // T*512 f32  2M
    unsigned short* qbuf  = (unsigned short*)(ws + 14 * MB);  // 4M
    unsigned short* kbuf  = (unsigned short*)(ws + 18 * MB);  // 1M
    unsigned short* vbuf  = (unsigned short*)(ws + 19 * MB);  // 1M
    unsigned short* attnb = (unsigned short*)(ws + 20 * MB);  // 4M
    float*          xnf   = (float*)         (ws + 24 * MB);  // 8M
    unsigned short* xnb   = (unsigned short*)(ws + 32 * MB);  // 4M
    float*          fullw = (float*)         (ws + 36 * MB);  // 32K
    float*          hgate = (float*)         (ws + 37 * MB);  // T*I f32 4M
    unsigned short* actb  = (unsigned short*)(ws + 41 * MB);  // 2M
    float*          moe   = (float*)         (ws + 43 * MB);  // 8M
    float*          ggate = (float*)         (ws + 51 * MB);  // T*SI f32 16M
    unsigned short* sactb = (unsigned short*)(ws + 67 * MB);  // 8M
    (void)ws_size; (void)in_sizes; (void)n_in; (void)out_size;

    // 1) RMSNorm 1 -> bf16 activations.
    rmsnorm_kernel<<<T, 256, 0, stream>>>(hidden, ln1, nullptr, xb);

    // 2) QKV projections.
    launch_gemm(stream, xb, D, wq, H * HD, T, H * HD, D,
                qf, nullptr, H * HD, GEMM_STORE, 0.f, nullptr, 0, nullptr, 0, nullptr, 0);
    launch_gemm(stream, xb, D, wk, KV * HD, T, KV * HD, D,
                kf, nullptr, KV * HD, GEMM_STORE, 0.f, nullptr, 0, nullptr, 0, nullptr, 0);
    launch_gemm(stream, xb, D, wv, KV * HD, T, KV * HD, D,
                nullptr, vbuf, KV * HD, GEMM_STORE, 0.f, nullptr, 0, nullptr, 0, nullptr, 0);

    // 3) RoPE on Q and K (f32 -> bf16).
    rope_kernel<<<(T * H  * 32 + 255) / 256, 256, 0, stream>>>(pos, qf, qbuf, H);
    rope_kernel<<<(T * KV * 32 + 255) / 256, 256, 0, stream>>>(pos, kf, kbuf, KV);

    // 4) Causal GQA attention.
    attn_kernel<<<dim3(H, T / 64), 128, 0, stream>>>(qbuf, kbuf, vbuf, attnb);

    // 5) Output projection + residual -> residOut (second output).
    launch_gemm(stream, attnb, H * HD, wo, D, T, D, H * HD,
                residOut, nullptr, D, GEMM_RESID, RES_MULT, hidden, D,
                nullptr, 0, nullptr, 0);

    // 6) RMSNorm 2.
    rmsnorm_kernel<<<T, 256, 0, stream>>>(residOut, ln2, xnf, xnb);

    // 7) Router top-2.
    router_kernel<<<T, 32, 0, stream>>>(xnf, routerw, fullw);

    // 8) Experts (dense, weighted by full_w as in the reference).
    for (int e = 0; e < E; ++e) {
        const float* w13e = w13 + (size_t)e * D * 2 * I;
        const float* w2e  = w2  + (size_t)e * I * D;
        launch_gemm(stream, xnb, D, w13e, 2 * I, T, I, D,
                    hgate, nullptr, I, GEMM_STORE, 0.f, nullptr, 0, nullptr, 0, nullptr, 0);
        launch_gemm(stream, xnb, D, w13e + I, 2 * I, T, I, D,
                    nullptr, actb, I, GEMM_SILU, 0.f, nullptr, 0, nullptr, 0, hgate, I);
        launch_gemm(stream, actb, I, w2e, D, T, D, I,
                    moe, nullptr, D, (e == 0) ? GEMM_MOE_INIT : GEMM_MOE_ACC, 0.f,
                    nullptr, 0, fullw + e, E, nullptr, 0);
    }

    // 9) Shared expert.
    launch_gemm(stream, xnb, D, sw13, 2 * SI, T, SI, D,
                ggate, nullptr, SI, GEMM_STORE, 0.f, nullptr, 0, nullptr, 0, nullptr, 0);
    launch_gemm(stream, xnb, D, sw13 + SI, 2 * SI, T, SI, D,
                nullptr, sactb, SI, GEMM_SILU, 0.f, nullptr, 0, nullptr, 0, ggate, SI);

    // 10) Shared down-proj fused with final residual.
    launch_gemm(stream, sactb, SI, sw2, D, T, D, SI,
                outF, nullptr, D, GEMM_FINAL, RES_MULT, residOut, D,
                nullptr, 0, moe, D);
}
